// BL_36721970381090
// MI455X (gfx1250) — matrix operand order repaired and checked by reading, not compiled, not run on hardware
//
#include <hip/hip_runtime.h>

typedef _Float16 h16;
typedef __attribute__((ext_vector_type(2)))  _Float16 h2;
typedef __attribute__((ext_vector_type(16))) _Float16 v16h;
typedef __attribute__((ext_vector_type(8)))  _Float16 v8h;
typedef __attribute__((ext_vector_type(8)))  float    v8f;

static __device__ __forceinline__ h2 cvt_pk_h2(float a, float b) {
    return __builtin_bit_cast(h2, __builtin_amdgcn_cvt_pkrtz(a, b));
}

static constexpr int kB  = 131072;
static constexpr int kD1 = 40;    // features
static constexpr int kD2 = 10;    // timesteps
static constexpr int kT0 = 120;
static constexpr int kT1 = 5;
static constexpr int kO0 = 3;

static constexpr int WAVES  = 4;
static constexpr int BPW    = 16;           // batch rows per wave (WMMA M)
static constexpr int BT     = WAVES * BPW;  // 64 batch rows per workgroup
static constexpr int KW     = 64;           // padded K for W11 in LDS
static constexpr int KY     = 48;           // padded K for Y in LDS
static constexpr int NTILES = 8;            // ceil(120/16) t-tiles

__global__ __launch_bounds__(WAVES * 32)
void fused_mlp_wmma(const float* __restrict__ x,
                    const float* __restrict__ W11,
                    const float* __restrict__ fc2w,
                    const float* __restrict__ bias1,
                    const float* __restrict__ W12,
                    const float* __restrict__ fc4w,
                    const float* __restrict__ bias2,
                    float* __restrict__ out)
{
    __shared__ h16   sW11[128 * KW];                 // [t][d], zero padded f16
    __shared__ h16   sY[WAVES][kT1][BPW * KY];       // per wave: [u][b_local][d]
    __shared__ float sK[kT1][kO0][128];              // W12[o,t]*fc4[u], zero padded
    __shared__ float sB1[128 * kT1];                 // bias1[t,u], zero padded
    __shared__ float sRed[WAVES][BPW * kO0];         // per-wave output accumulators

    const int tid   = threadIdx.x;
    const int lane  = tid & 31;
    const int wv    = tid >> 5;
    const int bBase = blockIdx.x * BT + wv * BPW;

    const float* xw = x + (long)bBase * (kD1 * kD2);

    // ---- prefetch this wave's x block (overlaps LDS setup) ----
    for (int i = lane; i < (BPW * kD1 * kD2 * 4) / 128; i += 32)
        __builtin_prefetch(((const char*)xw) + i * 128, 0, 3);

    // ---- W11 -> LDS as f16, zero padded to 128x64 (cooperative) ----
    for (int i = tid; i < 128 * KW; i += WAVES * 32) {
        const int t = i / KW, d = i % KW;
        sW11[i] = (t < kT0 && d < kD1) ? (h16)W11[t * kD1 + d] : (h16)0.0f;
    }
    // ---- coefficient table: sK[u][o][t] = W12[o,t] * fc4[u], zero padded ----
    for (int i = tid; i < kT1 * kO0 * 128; i += WAVES * 32) {
        const int u = i / (kO0 * 128);
        const int o = (i / 128) % kO0;
        const int t = i % 128;
        sK[u][o][t] = (t < kT0) ? W12[o * kT0 + t] * fc4w[u] : 0.0f;
    }
    // ---- bias table: sB1[t*5+u], zero padded ----
    for (int i = tid; i < 128 * kT1; i += WAVES * 32) {
        const int t = i / kT1;
        sB1[i] = (t < kT0) ? bias1[i] : 0.0f;
    }
    // ---- zero this wave's Y staging region (covers K padding) + reducers ----
    {
        unsigned* z = (unsigned*)&sY[wv][0][0];
        for (int i = lane; i < (kT1 * BPW * KY) / 2; i += 32) z[i] = 0u;
        float* zr = &sRed[wv][0];
        for (int i = lane; i < BPW * kO0; i += 32) zr[i] = 0.0f;
    }

    // ---- stage 1: y[b,d,u] = sum_s x[b,d,s] * fc2w[u,s]  (packed f16) ----
    h2 f2p[kT1][kD2 / 2];
    #pragma unroll
    for (int u = 0; u < kT1; ++u)
        #pragma unroll
        for (int j = 0; j < kD2 / 2; ++j)
            f2p[u][j] = cvt_pk_h2(fc2w[u * kD2 + 2 * j],
                                  fc2w[u * kD2 + 2 * j + 1]);

    #pragma unroll
    for (int it = 0; it < (BPW * kD1) / 32; ++it) {   // 20 iterations
        const int p  = it * 32 + lane;                // (b_local, d), contiguous
        const int bl = p / kD1;
        const int d  = p % kD1;
        const float2* xr = (const float2*)(xw + p * kD2);  // 8B aligned
        h2 xh[kD2 / 2];
        #pragma unroll
        for (int j = 0; j < kD2 / 2; ++j) {
            const float2 v = xr[j];
            xh[j] = cvt_pk_h2(v.x, v.y);
        }
        #pragma unroll
        for (int u = 0; u < kT1; ++u) {
            h2 a = xh[0] * f2p[u][0];
            #pragma unroll
            for (int j = 1; j < kD2 / 2; ++j) a += xh[j] * f2p[u][j];
            sY[wv][u][bl * KY + d] = a.x + a.y;
        }
    }

    __syncthreads();

    // ---- stage 2: P[b,t,u] via WMMA + fused relu-epilogue ----
    const int nsub = lane & 15;
    const int c0 = (lane < 16) ? 0 : 8;    // A-operand K-chunk base (16-bit A layout)
    const int kb = (lane < 16) ? 0 : 8;    // B-operand K-chunk base (same layout as A)
    const v8h z8 = {};

    float acc[kO0][8];
    #pragma unroll
    for (int o = 0; o < kO0; ++o)
        #pragma unroll
        for (int r = 0; r < 8; ++r) acc[o][r] = 0.0f;

    for (int u = 0; u < kT1; ++u) {
        const h16* yrow = &sY[wv][u][0] + nsub * KY;  // row M = lane&15

        // A for K-step 0 (K=0..31): chunks at d=c0 and d=c0+16
        v8h a0lo = *(const v8h*)(yrow + c0);
        v8h a0hi = *(const v8h*)(yrow + c0 + 16);
        // A for K-step 1 (K=32..63): chunk1 at d=32+c0 (zeros past 40), chunk2 zero
        v8h a1lo = *(const v8h*)(yrow + 32 + c0);
        v16h A0 = __builtin_shufflevector(a0lo, a0hi, 0,1,2,3,4,5,6,7,8,9,10,11,12,13,14,15);
        v16h A1 = __builtin_shufflevector(a1lo, z8,   0,1,2,3,4,5,6,7,8,9,10,11,12,13,14,15);

        for (int nt = 0; nt < NTILES; ++nt) {
            const h16* wrow = sW11 + (nt * 16 + nsub) * KW;  // column N = lane&15
            v8h b0lo = *(const v8h*)(wrow + kb);
            v8h b0hi = *(const v8h*)(wrow + kb + 16);
            v8h b1lo = *(const v8h*)(wrow + 32 + kb);        // zero-padded past d=40
            v8h b1hi = *(const v8h*)(wrow + 48 + kb);
            v16h B0 = __builtin_shufflevector(b0lo, b0hi, 0,1,2,3,4,5,6,7,8,9,10,11,12,13,14,15);
            v16h B1 = __builtin_shufflevector(b1lo, b1hi, 0,1,2,3,4,5,6,7,8,9,10,11,12,13,14,15);

            v8f c = {};
            c = __builtin_amdgcn_wmma_f32_16x16x32_f16(false, A0, false, B0,
                                                       (short)0, c, false, false);
            c = __builtin_amdgcn_wmma_f32_16x16x32_f16(false, A1, false, B1,
                                                       (short)0, c, false, false);

            // fused epilogue: relu(P + bias1[t,u]) weighted by W12[o,t]*fc4[u]
            const int tl = nt * 16 + nsub;                 // padded index, no guard
            const float b1u = sB1[tl * kT1 + u];
            const float k0 = sK[u][0][tl];
            const float k1 = sK[u][1][tl];
            const float k2 = sK[u][2][tl];
            #pragma unroll
            for (int r = 0; r < 8; ++r) {
                float v = c[r] + b1u;
                v = (v > 0.0f) ? v : 0.0f;
                acc[0][r] += k0 * v;
                acc[1][r] += k1 * v;
                acc[2][r] += k2 * v;
            }
        }
    }

    // ---- reduction over t (lanes of each half) via LDS float atomics ----
    const int rowBase = (lane >> 4) * 8;   // C layout: rows 0-7 lanes<16, 8-15 lanes>=16
    #pragma unroll
    for (int r = 0; r < 8; ++r)
        #pragma unroll
        for (int o = 0; o < kO0; ++o)
            atomicAdd(&sRed[wv][(rowBase + r) * kO0 + o], acc[o][r]);

    __syncthreads();

    if (lane < BPW) {
        const int b = bBase + lane;
        const float* s = &sRed[wv][lane * kO0];
        out[b * kO0 + 0] = s[0] + bias2[0];
        out[b * kO0 + 1] = s[1] + bias2[1];
        out[b * kO0 + 2] = s[2] + bias2[2];
    }
}

extern "C" void kernel_launch(void* const* d_in, const int* in_sizes, int n_in,
                              void* d_out, int out_size, void* d_ws, size_t ws_size,
                              hipStream_t stream) {
    (void)in_sizes; (void)n_in; (void)d_ws; (void)ws_size; (void)out_size;
    const float* x     = (const float*)d_in[0];
    const float* W11   = (const float*)d_in[1];
    const float* fc2w  = (const float*)d_in[2];
    const float* bias1 = (const float*)d_in[3];
    const float* W12   = (const float*)d_in[4];
    const float* fc4w  = (const float*)d_in[5];
    const float* bias2 = (const float*)d_in[6];
    float* out = (float*)d_out;

    dim3 grid(kB / BT);       // 2048 workgroups
    dim3 block(WAVES * 32);   // 128 threads = 4 waves
    fused_mlp_wmma<<<grid, block, 0, stream>>>(x, W11, fc2w, bias1, W12, fc4w, bias2, out);
}
